// CausalSelfAttention_78589311582661
// MI455X (gfx1250) — matrix operand order repaired and checked by reading, not compiled, not run on hardware
//
#include <hip/hip_runtime.h>

// Problem constants (from reference)
#define BATCH   4
#define SEQ     2048
#define EMBED   1024
#define NHEAD   16
#define HDIM    64
#define THREEC  (3*EMBED)          // 3072
#define MROWS   (BATCH*SEQ)        // 8192

typedef __bf16 bf16;
typedef __attribute__((ext_vector_type(16))) __bf16 v16bf;
typedef __attribute__((ext_vector_type(8)))  float  v8f;

union Frag {
  uint4 q[2];
  v16bf bf;
  bf16  h[16];
};

__device__ __forceinline__ v16bf load_frag16(const bf16* p0, const bf16* p1) {
  Frag u;
  u.q[0] = *(const uint4*)p0;
  u.q[1] = *(const uint4*)p1;
  return u.bf;
}

__device__ __forceinline__ v8f wmma_bf16(v16bf a, v16bf b, v8f c) {
  // D = A(16x32 bf16) * B(32x16 bf16) + C(16x16 f32)
  return __builtin_amdgcn_wmma_f32_16x16x32_bf16(
      false, a, false, b, (short)0, c, false, false);
}

// LDS byte offset of a shared-memory pointer (addrspace(3) value)
__device__ __forceinline__ unsigned lds_addr(const void* p) {
  return (unsigned)(unsigned long long)
         (__attribute__((address_space(3))) const void*)p;
}

// Async 16B copy global -> LDS (ASYNCcnt-tracked, no VGPR round trip)
__device__ __forceinline__ void async_copy_b128(unsigned lds_dst, const void* gsrc) {
  unsigned long long ga = (unsigned long long)gsrc;
  asm volatile("global_load_async_to_lds_b128 %0, %1, off"
               :: "v"(lds_dst), "v"(ga) : "memory");
}

__device__ __forceinline__ void wait_async0() {
  asm volatile("s_wait_asynccnt 0" ::: "memory");
}

// Butterfly max over 16 lanes via immediate-pattern ds_swizzle (SWAPX1/2/4/8)
__device__ __forceinline__ float redmax16(float v) {
  v = fmaxf(v, __int_as_float(__builtin_amdgcn_ds_swizzle(__float_as_int(v), 0x041F)));
  v = fmaxf(v, __int_as_float(__builtin_amdgcn_ds_swizzle(__float_as_int(v), 0x081F)));
  v = fmaxf(v, __int_as_float(__builtin_amdgcn_ds_swizzle(__float_as_int(v), 0x101F)));
  v = fmaxf(v, __int_as_float(__builtin_amdgcn_ds_swizzle(__float_as_int(v), 0x201F)));
  return v;
}

// ---------------------------------------------------------------------------
// Kernel 1: f32 -> bf16 conversion, 8 elements / thread (b128 in, b128 out)
// ---------------------------------------------------------------------------
__global__ __launch_bounds__(256) void cvt_f32_bf16(const float* __restrict__ src,
                                                    bf16* __restrict__ dst, int n8) {
  int i = blockIdx.x * blockDim.x + threadIdx.x;
  if (i >= n8) return;
  float4 a = ((const float4*)src)[2*i + 0];
  float4 b = ((const float4*)src)[2*i + 1];
  Frag u;
  u.h[0] = (bf16)a.x; u.h[1] = (bf16)a.y; u.h[2] = (bf16)a.z; u.h[3] = (bf16)a.w;
  u.h[4] = (bf16)b.x; u.h[5] = (bf16)b.y; u.h[6] = (bf16)b.z; u.h[7] = (bf16)b.w;
  ((uint4*)dst)[i] = u.q[0];
}

// ---------------------------------------------------------------------------
// Kernel 2/4: bf16 WMMA GEMM  C[M,N] = A[M,K] @ B[K,N] + bias[N]
// Block tile 128x256, 256 threads (8 waves, 2x4), wave tile 64x64
// (16 WMMAs per K-step vs 8 fragment loads). A staged via async-to-LDS.
// ---------------------------------------------------------------------------
template<bool OUT_F32>
__global__ __launch_bounds__(256) void gemm_bf16_wmma(
    const bf16* __restrict__ A, const bf16* __restrict__ Bm,
    const float* __restrict__ bias, void* __restrict__ Cout,
    int Ndim, int Kdim)
{
  __shared__ bf16 sA[128][48];   // [m][k]  row stride 96B (16B aligned)
  __shared__ bf16 sB[256][48];   // [n][k]  K-contiguous for B fragments

  const int tid  = threadIdx.x;
  const int lane = tid & 31;
  const int wave = tid >> 5;
  const int lo   = lane & 15;
  const int hi   = lane >> 4;
  const int wm   = wave & 1;     // 2 waves along M (64 rows each)
  const int wn   = wave >> 1;    // 4 waves along N (64 cols each)
  const int m0   = blockIdx.y * 128;
  const int n0   = blockIdx.x * 256;

  const v8f vz = {0.f,0.f,0.f,0.f,0.f,0.f,0.f,0.f};
  v8f acc[4][4];
  #pragma unroll
  for (int i = 0; i < 4; ++i)
    #pragma unroll
    for (int j = 0; j < 4; ++j) acc[i][j] = vz;

  const int arow = tid >> 1, ahalf = tid & 1;  // A staging: 2 threads/row (32B each)
  const int bkr  = tid >> 3, bseg  = tid & 7;  // B staging: 8 threads/k-row (64B each)

  const unsigned ldsA0 = lds_addr(&sA[arow][ahalf*16]);
  const unsigned ldsA1 = lds_addr(&sA[arow][ahalf*16 + 8]);

  for (int kk = 0; kk < Kdim; kk += 32) {
    // prefetch next K-chunk of the streamed operands into L2/WGP$
    if (kk + 32 < Kdim) {
      __builtin_prefetch(A  + (size_t)(m0 + arow) * Kdim + kk + 32 + ahalf*16, 0, 1);
      __builtin_prefetch(Bm + (size_t)(kk + 32 + bkr) * Ndim + n0 + bseg*32,   0, 1);
    }
    // stage A tile (contiguous along K) via async copy
    {
      const bf16* src = A + (size_t)(m0 + arow) * Kdim + kk + ahalf * 16;
      async_copy_b128(ldsA0, src);
      async_copy_b128(ldsA1, src + 8);
    }
    // stage B tile transposed -> [n][k] (manual: needs transpose)
    {
      const bf16* src = Bm + (size_t)(kk + bkr) * Ndim + n0 + bseg * 32;
      Frag u0, u1;
      u0.q[0] = ((const uint4*)src)[0];
      u0.q[1] = ((const uint4*)src)[1];
      u1.q[0] = ((const uint4*)src)[2];
      u1.q[1] = ((const uint4*)src)[3];
      #pragma unroll
      for (int e = 0; e < 16; ++e) sB[bseg*32 + e][bkr]      = u0.h[e];
      #pragma unroll
      for (int e = 0; e < 16; ++e) sB[bseg*32 + 16 + e][bkr] = u1.h[e];
    }
    wait_async0();
    __syncthreads();

    v16bf bfr[4];
    #pragma unroll
    for (int nf = 0; nf < 4; ++nf) {
      const int n = wn*64 + nf*16 + lo;
      bfr[nf] = load_frag16(&sB[n][hi*8], &sB[n][hi*8 + 16]);
    }
    #pragma unroll
    for (int mf = 0; mf < 4; ++mf) {
      const int m = wm*64 + mf*16 + lo;
      v16bf af = load_frag16(&sA[m][hi*8], &sA[m][hi*8 + 16]);
      #pragma unroll
      for (int nf = 0; nf < 4; ++nf)
        acc[mf][nf] = wmma_bf16(af, bfr[nf], acc[mf][nf]);
    }
    __syncthreads();
  }

  // epilogue: bias + store (f32 or bf16)
  #pragma unroll
  for (int mf = 0; mf < 4; ++mf) {
    #pragma unroll
    for (int nf = 0; nf < 4; ++nf) {
      const int n = n0 + wn*64 + nf*16 + lo;
      const float bv = bias[n];
      #pragma unroll
      for (int r = 0; r < 8; ++r) {
        const int row = m0 + wm*64 + mf*16 + r + 8*hi;
        const float val = acc[mf][nf][r] + bv;
        if (OUT_F32) ((float*)Cout)[(size_t)row * Ndim + n] = val;
        else         ((bf16*)Cout)[(size_t)row * Ndim + n] = (bf16)val;
      }
    }
  }
}

// ---------------------------------------------------------------------------
// Kernel 3: flash attention.  1 block = (b, h, 64 Q rows); 4 waves x 16 rows.
// 64-key chunks; K staged async; V staged transposed with an extra "ones"
// d-chunk so the softmax denominator accumulates as a 5th WMMA column
// (no cross-lane sum reduction; alpha-rescale updates it for free).
// ---------------------------------------------------------------------------
__global__ __launch_bounds__(128) void attn_flash_wmma(
    const bf16* __restrict__ qkv, const int* __restrict__ amask,
    bf16* __restrict__ y)
{
  __shared__ bf16 kt[64][72];      // K chunk [key][d]
  __shared__ bf16 vt[80][72];      // V chunk transposed [d][key]; rows 64..79 = 1.0
  __shared__ bf16 ps[4][16][80];   // per-wave P relayout buffer (16 x 64)
  __shared__ int  smask[64];       // attention_mask chunk

  const int tid  = threadIdx.x;
  const int lane = tid & 31;
  const int w    = tid >> 5;
  const int lo   = lane & 15;
  const int hi   = lane >> 4;

  const int ntile = SEQ / 64;                      // 32
  const int mt = blockIdx.x % ntile;
  const int h  = (blockIdx.x / ntile) % NHEAD;
  const int b  = blockIdx.x / (ntile * NHEAD);
  const int m0  = mt * 64;
  const int m0w = m0 + w * 16;

  const float LOG2E = 1.4426950408889634f;
  const float SCALE = 0.125f;                      // 1/sqrt(64)

  // ones rows for the denominator trick (read only after first barrier)
  for (int idx = tid; idx < 16*72; idx += 128)
    vt[64 + idx/72][idx%72] = (bf16)1.0f;

  // Q fragments (registers, reused across all key chunks)
  v16bf qf[2];
  {
    const bf16* base = qkv + (size_t)(b*SEQ + m0w + lo) * THREEC + h * HDIM;
    #pragma unroll
    for (int c = 0; c < 2; ++c) {
      const int d0 = c*32 + hi*8;
      qf[c] = load_frag16(base + d0, base + d0 + 16);
    }
  }

  const v8f vz = {0.f,0.f,0.f,0.f,0.f,0.f,0.f,0.f};
  v8f oacc[5];                                     // [0..3]=O, [4]=row-sum l
  #pragma unroll
  for (int i = 0; i < 5; ++i) oacc[i] = vz;
  float mrow[8];
  #pragma unroll
  for (int r = 0; r < 8; ++r) mrow[r] = -__builtin_inff();

  const int nchunks = m0/64 + 1;                   // causal: keys <= m0+63
  const int skey = tid >> 1, sseg = tid & 1;       // staging map (64 rows x 2)
  const int d0s  = sseg * 32;

  unsigned ldsK[4];
  #pragma unroll
  for (int q = 0; q < 4; ++q) ldsK[q] = lds_addr(&kt[skey][d0s + q*8]);

  for (int jc = 0; jc < nchunks; ++jc) {
    const int j = jc * 64;
    // stage K chunk [64][64] via async copy (4 x b128 per thread)
    {
      const bf16* src = qkv + (size_t)(b*SEQ + j + skey) * THREEC
                        + EMBED + h*HDIM + d0s;
      #pragma unroll
      for (int q = 0; q < 4; ++q) async_copy_b128(ldsK[q], src + q*8);
    }
    // stage V chunk transposed -> [d][key]
    {
      const bf16* src = qkv + (size_t)(b*SEQ + j + skey) * THREEC
                        + 2*EMBED + h*HDIM + d0s;
      Frag u0, u1;
      u0.q[0] = ((const uint4*)src)[0];
      u0.q[1] = ((const uint4*)src)[1];
      u1.q[0] = ((const uint4*)src)[2];
      u1.q[1] = ((const uint4*)src)[3];
      #pragma unroll
      for (int e = 0; e < 16; ++e) vt[d0s + e][skey]      = u0.h[e];
      #pragma unroll
      for (int e = 0; e < 16; ++e) vt[d0s + 16 + e][skey] = u1.h[e];
    }
    // stage attention_mask chunk
    if (tid < 64) smask[tid] = amask[b*SEQ + j + tid];
    wait_async0();
    __syncthreads();

    // S = Q @ K^T  (four 16-key sub-tiles)
    v8f sf[4];
    #pragma unroll
    for (int kf = 0; kf < 4; ++kf) {
      v8f s = vz;
      #pragma unroll
      for (int c = 0; c < 2; ++c) {
        const int keyr = kf*16 + lo;
        const int d0 = c*32 + hi*8;
        v16bf kfr = load_frag16(&kt[keyr][d0], &kt[keyr][d0 + 16]);
        s = wmma_bf16(qf[c], kfr, s);
      }
      sf[kf] = s;
    }

    // scale + causal & attention mask (mask from LDS)
    #pragma unroll
    for (int kf = 0; kf < 4; ++kf) {
      const int key = j + kf*16 + lo;
      const int av  = smask[kf*16 + lo];
      #pragma unroll
      for (int r = 0; r < 8; ++r) {
        const int row = m0w + r + 8*hi;
        const float sv = sf[kf][r] * SCALE;
        sf[kf][r] = (av != 0 && key <= row) ? sv : -__builtin_inff();
      }
    }

    // online softmax: row max (ds_swizzle butterfly), alpha, exp
    float alpha[8];
    #pragma unroll
    for (int r = 0; r < 8; ++r) {
      float v = fmaxf(fmaxf(sf[0][r], sf[1][r]), fmaxf(sf[2][r], sf[3][r]));
      v = redmax16(v);
      const float mnew = fmaxf(mrow[r], v);
      alpha[r] = exp2f((mrow[r] - mnew) * LOG2E);
      mrow[r] = mnew;
    }
    #pragma unroll
    for (int kf = 0; kf < 4; ++kf)
      #pragma unroll
      for (int r = 0; r < 8; ++r)
        sf[kf][r] = exp2f((sf[kf][r] - mrow[r]) * LOG2E);
    #pragma unroll
    for (int df = 0; df < 5; ++df)          // includes the l accumulator
      #pragma unroll
      for (int r = 0; r < 8; ++r)
        oacc[df][r] *= alpha[r];

    // relayout P (C/D layout) -> A-fragment layout via per-wave LDS
    #pragma unroll
    for (int kf = 0; kf < 4; ++kf)
      #pragma unroll
      for (int r = 0; r < 8; ++r)
        ps[w][r + 8*hi][kf*16 + lo] = (bf16)sf[kf][r];
    asm volatile("s_wait_dscnt 0" ::: "memory");   // wave-local LDS RAW
    v16bf pf[2];
    #pragma unroll
    for (int af = 0; af < 2; ++af)
      pf[af] = load_frag16(&ps[w][lo][af*32 + hi*8],
                           &ps[w][lo][af*32 + hi*8 + 16]);

    // O += P @ V  (4 d-chunks + 1 ones-chunk, 2 key-halves each)
    #pragma unroll
    for (int df = 0; df < 5; ++df) {
      #pragma unroll
      for (int af = 0; af < 2; ++af) {
        v16bf vfr = load_frag16(&vt[df*16 + lo][af*32 + hi*8],
                                &vt[df*16 + lo][af*32 + hi*8 + 16]);
        oacc[df] = wmma_bf16(pf[af], vfr, oacc[df]);
      }
    }
    __syncthreads();
  }

  // normalize by l (= oacc[4], identical across columns) and store y (bf16)
  float inv[8];
  #pragma unroll
  for (int r = 0; r < 8; ++r) inv[r] = 1.0f / oacc[4][r];
  #pragma unroll
  for (int df = 0; df < 4; ++df) {
    #pragma unroll
    for (int r = 0; r < 8; ++r) {
      const int row = m0w + r + 8*hi;
      y[(size_t)(b*SEQ + row) * EMBED + h*HDIM + df*16 + lo] =
          (bf16)(oacc[df][r] * inv[r]);
    }
  }
}

// ---------------------------------------------------------------------------
// Launcher. Workspace layout (bytes), ~92 MB total:
//   xb   bf16[8192*1024]   @ 0
//   Wab  bf16[1024*3072]   @ 16777216
//   Wpb  bf16[1024*1024]   @ 23068672
//   qkvb bf16[8192*3072]   @ 25165824
//   yb   bf16[8192*1024]   @ 75497472
// ---------------------------------------------------------------------------
extern "C" void kernel_launch(void* const* d_in, const int* in_sizes, int n_in,
                              void* d_out, int out_size, void* d_ws, size_t ws_size,
                              hipStream_t stream) {
  const float* x      = (const float*)d_in[0];
  const int*   amask  = (const int*)  d_in[1];
  const float* W_attn = (const float*)d_in[2];
  const float* b_attn = (const float*)d_in[3];
  const float* W_proj = (const float*)d_in[4];
  const float* b_proj = (const float*)d_in[5];

  char* ws = (char*)d_ws;
  bf16* xb   = (bf16*)(ws + 0);
  bf16* Wab  = (bf16*)(ws + 16777216);
  bf16* Wpb  = (bf16*)(ws + 23068672);
  bf16* qkvb = (bf16*)(ws + 25165824);
  bf16* yb   = (bf16*)(ws + 75497472);

  // 1) convert inputs to bf16
  cvt_f32_bf16<<<(MROWS*EMBED/8 + 255)/256, 256, 0, stream>>>(x, xb, MROWS*EMBED/8);
  cvt_f32_bf16<<<(EMBED*THREEC/8 + 255)/256, 256, 0, stream>>>(W_attn, Wab, EMBED*THREEC/8);
  cvt_f32_bf16<<<(EMBED*EMBED/8 + 255)/256, 256, 0, stream>>>(W_proj, Wpb, EMBED*EMBED/8);

  // 2) qkv = x @ W_attn + b_attn   (bf16 out)
  gemm_bf16_wmma<false><<<dim3(THREEC/256, MROWS/128), 256, 0, stream>>>(
      xb, Wab, b_attn, (void*)qkvb, THREEC, EMBED);

  // 3) flash attention -> yb (bf16)
  attn_flash_wmma<<<BATCH * NHEAD * (SEQ/64), 128, 0, stream>>>(qkvb, amask, yb);

  // 4) out = yb @ W_proj + b_proj  (f32 out)
  gemm_bf16_wmma<true><<<dim3(EMBED/256, MROWS/128), 256, 0, stream>>>(
      yb, Wpb, b_proj, d_out, EMBED, EMBED);
}
